// RCNN_91216515433179
// MI455X (gfx1250) — compile-verified, matrix operand order checked by reading, not run on hardware
//
#include <hip/hip_runtime.h>
#include <hip/hip_bf16.h>

typedef __attribute__((ext_vector_type(16))) _Float16 v16h;
typedef __attribute__((ext_vector_type(8)))  float    v8f;

#define BATCH 8
#define NN    4096
#define HH    64

// ---------------------------------------------------------------------------
// CDNA5 async global->LDS path (ASYNCcnt-tracked DMA, no VGPR staging).
// Probe-confirmed signature: (v4i __device__* src, v4i __shared__* dst,
//                             imm int offset, imm int cpol)
// ---------------------------------------------------------------------------
#if __has_builtin(__builtin_amdgcn_global_load_async_to_lds_b128) && \
    __has_builtin(__builtin_amdgcn_s_wait_asynccnt)
#define USE_ASYNC_LDS 1
#pragma message("CDNA5: async global->LDS path ENABLED")
#else
#define USE_ASYNC_LDS 0
#pragma message("CDNA5: async global->LDS path DISABLED (builtins missing)")
#endif

typedef int v4i_g __attribute__((vector_size(4 * sizeof(int))));
typedef __attribute__((address_space(1))) v4i_g* g_v4i_ptr;
typedef __attribute__((address_space(3))) v4i_g* s_v4i_ptr;

__device__ __forceinline__ void copy16_to_lds(const float* g, float* l) {
#if USE_ASYNC_LDS
    __builtin_amdgcn_global_load_async_to_lds_b128((g_v4i_ptr)g, (s_v4i_ptr)l, 0, 0);
#else
    float4 v = *(const float4*)g;
    *(float4*)l = v;
#endif
}

// ---------------------------------------------------------------------------
// GEMM 1: Y[row, h] = sum_f X[row, f] * W[h, f] + bias[h]    (rows = B*N, K=64)
// One block = 128 threads = 4 waves; computes 16 rows x 64 cols via WMMA f16.
// ---------------------------------------------------------------------------
__global__ void xw_gemm_kernel(const float* __restrict__ X,
                               const float* __restrict__ W,   // (64 out, 64 in)
                               const float* __restrict__ bias,
                               float* __restrict__ out) {
    __shared__ _Float16 Ah[16][68];   // 16 rows x 64 K (padded)
    __shared__ _Float16 Wh[64][68];   // [k][n] = W[n][k]  (padded)

    const int tid  = threadIdx.x;
    const int wave = tid >> 5;
    const int lane = tid & 31;
    const int hv   = lane >> 4;       // half: 0 or 1
    const int row  = lane & 15;
    const int m0   = blockIdx.x * 16;

    for (int i = tid; i < 16 * 64; i += 128) {
        int r = i >> 6, f = i & 63;
        Ah[r][f] = (_Float16)X[(size_t)(m0 + r) * 64 + f];
    }
    for (int i = tid; i < 64 * 64; i += 128) {
        int n = i >> 6, k = i & 63;
        Wh[k][n] = (_Float16)W[i];
    }
    __syncthreads();

    v8f c = {};
    #pragma unroll
    for (int kk0 = 0; kk0 < 64; kk0 += 32) {
        v16h a, b;
        #pragma unroll
        for (int j = 0; j < 8; ++j) {
            a[j]     = Ah[row][kk0 + hv * 8 + j];
            a[j + 8] = Ah[row][kk0 + hv * 8 + 16 + j];
            b[j]     = Wh[kk0 + hv * 8 + j][wave * 16 + row];
            b[j + 8] = Wh[kk0 + hv * 8 + 16 + j][wave * 16 + row];
        }
        c = __builtin_amdgcn_wmma_f32_16x16x32_f16(false, a, false, b,
                                                   (short)0, c, false, false);
    }

    #pragma unroll
    for (int rr = 0; rr < 8; ++rr) {
        int m = hv * 8 + rr;
        int n = wave * 16 + row;
        out[(size_t)(m0 + m) * 64 + n] = c[rr] + bias[n];
    }
}

// ---------------------------------------------------------------------------
// GEMM 2: Z[b] = adj[b] (4096x4096) @ Y[b] (4096x64).   HBM-bound on adj.
// Double-buffered async-to-LDS pipeline: each K-step issues 5 async b128
// instructions for tile k+1, waits asynccnt<=5 (previous group done), then
// WMMAs the current buffer. f32->f16 conversion happens in the gather (VALU),
// overlapping the matrix pipe.
// ---------------------------------------------------------------------------
__global__ void adj_gemm_kernel(const float* __restrict__ adj,
                                const float* __restrict__ Y,
                                float* __restrict__ out) {
    __shared__ __align__(16) float Af[2][16][36];   // row stride 144B (16B mult)
    __shared__ __align__(16) float Bf[2][32][68];   // row stride 272B (16B mult)

    const int b    = blockIdx.y;
    const int m0   = blockIdx.x * 16;
    const int tid  = threadIdx.x;
    const int wave = tid >> 5;
    const int lane = tid & 31;
    const int hv   = lane >> 4;
    const int row  = lane & 15;

    const float* Arow = adj + ((size_t)b * NN + m0) * NN;
    const float* Bmat = Y + (size_t)b * NN * 64;

    auto issue = [&](int buf, int k0) {
        // A tile: 16 rows x 32 K = 128 x 16B chunks -> 1 per thread
        {
            int r = tid >> 3, ch = tid & 7;
            copy16_to_lds(Arow + (size_t)r * NN + k0 + ch * 4, &Af[buf][r][ch * 4]);
            // stream hint for the adj rows a few tiles ahead (global_prefetch_b8)
            if (k0 + 256 < NN)
                __builtin_prefetch(Arow + (size_t)r * NN + k0 + 256 + ch * 4, 0, 0);
        }
        // B tile: 32 rows x 64 cols = 512 x 16B chunks -> 4 per thread
        #pragma unroll
        for (int it = 0; it < 4; ++it) {
            int idx = tid + 128 * it;
            int r = idx >> 4, ch = idx & 15;
            copy16_to_lds(Bmat + (size_t)(k0 + r) * 64 + ch * 4, &Bf[buf][r][ch * 4]);
        }
    };

    issue(0, 0);

    v8f c = {};
    for (int k0 = 0; k0 < NN; k0 += 32) {
        const int cur = (k0 >> 5) & 1;
        if (k0 + 32 < NN) {
            issue(cur ^ 1, k0 + 32);
#if USE_ASYNC_LDS
            __builtin_amdgcn_s_wait_asynccnt(5);   // previous group complete
#endif
        } else {
#if USE_ASYNC_LDS
            __builtin_amdgcn_s_wait_asynccnt(0);
#endif
        }
        __syncthreads();   // all waves' tiles visible

        v16h a, bb;
        #pragma unroll
        for (int j = 0; j < 8; ++j) {
            a[j]      = (_Float16)Af[cur][row][hv * 8 + j];
            a[j + 8]  = (_Float16)Af[cur][row][hv * 8 + 16 + j];
            bb[j]     = (_Float16)Bf[cur][hv * 8 + j][wave * 16 + row];
            bb[j + 8] = (_Float16)Bf[cur][hv * 8 + 16 + j][wave * 16 + row];
        }
        c = __builtin_amdgcn_wmma_f32_16x16x32_f16(false, a, false, bb,
                                                   (short)0, c, false, false);
        __syncthreads();   // reads done before next issue targets this buffer
    }

    #pragma unroll
    for (int rr = 0; rr < 8; ++rr) {
        int m = hv * 8 + rr;
        int n = wave * 16 + row;
        out[((size_t)b * NN + m0 + m) * 64 + n] = c[rr];
    }
}

// ---------------------------------------------------------------------------
// BatchNorm over channels (last dim = 64) across B*N rows.
// ---------------------------------------------------------------------------
__global__ void bn_stats_kernel(const float* __restrict__ x,
                                float* __restrict__ sums,  // [0:64]=sum, [64:128]=sumsq
                                int rows) {
    __shared__ float s1[64], s2[64];
    const int tid = threadIdx.x;           // 256
    if (tid < 64) { s1[tid] = 0.f; s2[tid] = 0.f; }
    __syncthreads();

    const int c    = tid & 63;
    const int roff = tid >> 6;             // 0..3
    const int chunk = (rows + gridDim.x - 1) / gridDim.x;
    const int r0 = blockIdx.x * chunk;
    const int r1 = min(r0 + chunk, rows);

    float a1 = 0.f, a2 = 0.f;
    for (int r = r0 + roff; r < r1; r += 4) {
        float v = x[(size_t)r * 64 + c];
        a1 += v; a2 += v * v;
    }
    atomicAdd(&s1[c], a1);
    atomicAdd(&s2[c], a2);
    __syncthreads();
    if (tid < 64) {
        atomicAdd(&sums[tid], s1[tid]);
        atomicAdd(&sums[64 + tid], s2[tid]);
    }
}

__global__ void bn_finalize_kernel(const float* __restrict__ sums,
                                   const float* __restrict__ gamma,
                                   const float* __restrict__ beta,
                                   float* __restrict__ ss, float inv_n) {
    int c = threadIdx.x;                   // 64
    float mean = sums[c] * inv_n;
    float var  = sums[64 + c] * inv_n - mean * mean;
    float sc   = gamma[c] * rsqrtf(var + 1e-5f);
    ss[c]      = sc;
    ss[64 + c] = beta[c] - mean * sc;
}

__global__ void bn_relu_kernel(float* __restrict__ x,
                               const float* __restrict__ ss, size_t n) {
    size_t i = (size_t)blockIdx.x * blockDim.x + threadIdx.x;
    if (i < n) {
        int c = (int)(i & 63);
        float v = x[i] * ss[c] + ss[64 + c];
        x[i] = fmaxf(v, 0.f);
    }
}

// ---------------------------------------------------------------------------
// 1-D "same" conv over time, data layout (b, t, c).
// ---------------------------------------------------------------------------
template <int K>
__global__ void conv1d_kernel(const float* __restrict__ in,
                              const float* __restrict__ w,     // (64,64,K)
                              const float* __restrict__ bias,
                              float* __restrict__ out) {
    const int P = K / 2;
    __shared__ float xin[4 + K - 1][64];

    const int b  = blockIdx.y;
    const int t0 = blockIdx.x * 4;
    const int tid = threadIdx.x;

    for (int i = tid; i < (4 + K - 1) * 64; i += 256) {
        int rr = i >> 6, c = i & 63;
        int t = t0 - P + rr;
        xin[rr][c] = (t >= 0 && t < NN) ? in[((size_t)b * NN + t) * 64 + c] : 0.f;
    }
    __syncthreads();

    const int co = tid & 63;
    const int tl = tid >> 6;               // 0..3
    float acc = bias[co];
    #pragma unroll
    for (int j = 0; j < K; ++j) {
        #pragma unroll 8
        for (int ci = 0; ci < 64; ++ci)
            acc = fmaf(w[(co * 64 + ci) * K + j], xin[tl + j][ci], acc);
    }
    out[((size_t)b * NN + t0 + tl) * 64 + co] = acc;
}

// ---------------------------------------------------------------------------
// LSTM layer: one persistent block per batch; 256 threads = one gate row each.
// Per-thread weight rows live in registers; x_t/h_t staged in LDS.
// ---------------------------------------------------------------------------
__global__ void lstm_layer_kernel(const float* __restrict__ x,
                                  const float* __restrict__ wih,  // (256,64)
                                  const float* __restrict__ whh,  // (256,64)
                                  const float* __restrict__ bih,
                                  const float* __restrict__ bhh,
                                  float* __restrict__ hout) {
    __shared__ float xs[64], hs[64], cs[64], gates[256];
    const int b   = blockIdx.x;
    const int tid = threadIdx.x;            // 256

    float rwi[64], rwh[64];
    {
        const float* wi = wih + (size_t)tid * 64;
        const float* wh = whh + (size_t)tid * 64;
        #pragma unroll
        for (int i = 0; i < 64; ++i) { rwi[i] = wi[i]; rwh[i] = wh[i]; }
    }
    const float bsum = bih[tid] + bhh[tid];
    if (tid < 64) { hs[tid] = 0.f; cs[tid] = 0.f; }
    __syncthreads();

    for (int t = 0; t < NN; ++t) {
        if (tid < 64) xs[tid] = x[((size_t)b * NN + t) * 64 + tid];
        __syncthreads();

        float acc = bsum;
        #pragma unroll
        for (int i = 0; i < 64; ++i) acc = fmaf(rwi[i], xs[i], acc);
        #pragma unroll
        for (int i = 0; i < 64; ++i) acc = fmaf(rwh[i], hs[i], acc);
        gates[tid] = acc;
        __syncthreads();

        if (tid < 64) {
            float ig = 1.f / (1.f + __expf(-gates[tid]));
            float fg = 1.f / (1.f + __expf(-gates[64 + tid]));
            float gg = tanhf(gates[128 + tid]);
            float og = 1.f / (1.f + __expf(-gates[192 + tid]));
            float cn = fg * cs[tid] + ig * gg;
            float hn = og * tanhf(cn);
            cs[tid] = cn; hs[tid] = hn;
            hout[((size_t)b * NN + t) * 64 + tid] = hn;
        }
        __syncthreads();
    }
}

// ---------------------------------------------------------------------------
// Attention logits, softmax over T, and final FC head.
// ---------------------------------------------------------------------------
__global__ void attn_logits_kernel(const float* __restrict__ h2,
                                   const float* __restrict__ w,   // (1,64)
                                   const float* __restrict__ b,
                                   float* __restrict__ logits) {
    __shared__ float sw[64];
    if (threadIdx.x < 64) sw[threadIdx.x] = w[threadIdx.x];
    __syncthreads();
    int idx = blockIdx.x * blockDim.x + threadIdx.x;   // B*N
    const float* hr = h2 + (size_t)idx * 64;
    float acc = b[0];
    #pragma unroll
    for (int c = 0; c < 64; ++c) acc = fmaf(hr[c], sw[c], acc);
    logits[idx] = acc;
}

__global__ void softmax_kernel(const float* __restrict__ logits,
                               float* __restrict__ attn) {
    __shared__ float red[256];
    const int b = blockIdx.x, tid = threadIdx.x;
    const float* L = logits + (size_t)b * NN;

    float m = -1e30f;
    for (int t = tid; t < NN; t += 256) m = fmaxf(m, L[t]);
    red[tid] = m; __syncthreads();
    for (int s = 128; s > 0; s >>= 1) {
        if (tid < s) red[tid] = fmaxf(red[tid], red[tid + s]);
        __syncthreads();
    }
    m = red[0];
    __syncthreads();

    float sum = 0.f;
    for (int t = tid; t < NN; t += 256) sum += __expf(L[t] - m);
    red[tid] = sum; __syncthreads();
    for (int s = 128; s > 0; s >>= 1) {
        if (tid < s) red[tid] += red[tid + s];
        __syncthreads();
    }
    float inv = 1.f / red[0];
    for (int t = tid; t < NN; t += 256)
        attn[(size_t)b * NN + t] = __expf(L[t] - m) * inv;
}

__global__ void head_kernel(const float* __restrict__ h2,
                            const float* __restrict__ attn,
                            const float* __restrict__ w1, const float* __restrict__ b1,
                            const float* __restrict__ w2, const float* __restrict__ b2,
                            const float* __restrict__ w3, const float* __restrict__ b3,
                            float* __restrict__ scores) {
    __shared__ float s1[32 * 64], sb1[32], s2[16 * 32], sb2[16], s3[16];
    const int tid = threadIdx.x;
    for (int i = tid; i < 32 * 64; i += 256) s1[i] = w1[i];
    for (int i = tid; i < 16 * 32; i += 256) s2[i] = w2[i];
    if (tid < 32) sb1[tid] = b1[tid];
    if (tid < 16) { sb2[tid] = b2[tid]; s3[tid] = w3[tid]; }
    __syncthreads();

    int idx = blockIdx.x * blockDim.x + tid;           // B*N
    float a = attn[idx];
    const float* hr = h2 + (size_t)idx * 64;
    float h[64];
    #pragma unroll
    for (int c = 0; c < 64; ++c) h[c] = hr[c] * a;

    float z1[32];
    #pragma unroll
    for (int k = 0; k < 32; ++k) {
        float acc = sb1[k];
        #pragma unroll 8
        for (int c = 0; c < 64; ++c) acc = fmaf(s1[k * 64 + c], h[c], acc);
        z1[k] = fmaxf(acc, 0.f);
    }
    float z2[16];
    #pragma unroll
    for (int k = 0; k < 16; ++k) {
        float acc = sb2[k];
        #pragma unroll
        for (int c = 0; c < 32; ++c) acc = fmaf(s2[k * 32 + c], z1[c], acc);
        z2[k] = fmaxf(acc, 0.f);
    }
    float s = b3[0];
    #pragma unroll
    for (int k = 0; k < 16; ++k) s = fmaf(s3[k], z2[k], s);
    scores[idx] = 2.f / (1.f + __expf(-s));
}

// ---------------------------------------------------------------------------
// Orchestration
// ---------------------------------------------------------------------------
extern "C" void kernel_launch(void* const* d_in, const int* in_sizes, int n_in,
                              void* d_out, int out_size, void* d_ws, size_t ws_size,
                              hipStream_t stream) {
    (void)in_sizes; (void)n_in; (void)out_size; (void)ws_size;

    const float* node_features = (const float*)d_in[0];
    const float* adj    = (const float*)d_in[1];
    const float* gc1_w  = (const float*)d_in[2];
    const float* gc1_b  = (const float*)d_in[3];
    const float* gc1_g  = (const float*)d_in[4];
    const float* gc1_be = (const float*)d_in[5];
    const float* gc2_w  = (const float*)d_in[6];
    const float* gc2_b  = (const float*)d_in[7];
    const float* gc2_g  = (const float*)d_in[8];
    const float* gc2_be = (const float*)d_in[9];
    const float* tc1_w  = (const float*)d_in[10];
    const float* tc1_b  = (const float*)d_in[11];
    const float* tc1_g  = (const float*)d_in[12];
    const float* tc1_be = (const float*)d_in[13];
    const float* tc2_w  = (const float*)d_in[14];
    const float* tc2_b  = (const float*)d_in[15];
    const float* tc2_g  = (const float*)d_in[16];
    const float* tc2_be = (const float*)d_in[17];
    const float* wih0   = (const float*)d_in[18];
    const float* whh0   = (const float*)d_in[19];
    const float* bih0   = (const float*)d_in[20];
    const float* bhh0   = (const float*)d_in[21];
    const float* wih1   = (const float*)d_in[22];
    const float* whh1   = (const float*)d_in[23];
    const float* bih1   = (const float*)d_in[24];
    const float* bhh1   = (const float*)d_in[25];
    const float* attn_w = (const float*)d_in[26];
    const float* attn_b = (const float*)d_in[27];
    const float* fc1_w  = (const float*)d_in[28];
    const float* fc1_b  = (const float*)d_in[29];
    const float* fc2_w  = (const float*)d_in[30];
    const float* fc2_b  = (const float*)d_in[31];
    const float* fc3_w  = (const float*)d_in[32];
    const float* fc3_b  = (const float*)d_in[33];

    const int rows = BATCH * NN;                     // 32768
    const size_t nfeat = (size_t)rows * 64;          // 2M floats

    float* bufA    = (float*)d_ws;
    float* bufB    = bufA + nfeat;
    float* logits  = bufB + nfeat;
    float* stats   = logits + rows;                  // 128 floats
    float* ss      = stats + 128;                    // 128 floats

    float* scores_out = (float*)d_out;               // [0 .. 32767]
    float* attn_out   = scores_out + rows;           // [32768 .. 65535]

    const dim3 blk128(128), blk256(256), blk64(64);
    const dim3 gGemmXW(rows / 16);                   // 2048
    const dim3 gGemmAdj(NN / 16, BATCH);             // 256 x 8
    const dim3 gStats(256);
    const dim3 gElem((unsigned)((nfeat + 255) / 256));
    const dim3 gConv(NN / 4, BATCH);
    const dim3 gRow(rows / 256);                     // 128
    const float inv_n = 1.0f / (float)rows;

    auto run_bn = [&](float* buf, const float* gamma, const float* beta) {
        (void)hipMemsetAsync(stats, 0, 128 * sizeof(float), stream);
        bn_stats_kernel<<<gStats, blk256, 0, stream>>>(buf, stats, rows);
        bn_finalize_kernel<<<1, blk64, 0, stream>>>(stats, gamma, beta, ss, inv_n);
        bn_relu_kernel<<<gElem, blk256, 0, stream>>>(buf, ss, nfeat);
    };

    // --- GC1 ---
    xw_gemm_kernel<<<gGemmXW, blk128, 0, stream>>>(node_features, gc1_w, gc1_b, bufA);
    adj_gemm_kernel<<<gGemmAdj, blk128, 0, stream>>>(adj, bufA, bufB);
    run_bn(bufB, gc1_g, gc1_be);

    // --- GC2 ---
    xw_gemm_kernel<<<gGemmXW, blk128, 0, stream>>>(bufB, gc2_w, gc2_b, bufA);
    adj_gemm_kernel<<<gGemmAdj, blk128, 0, stream>>>(adj, bufA, bufB);
    run_bn(bufB, gc2_g, gc2_be);

    // --- Temporal convs ---
    conv1d_kernel<3><<<gConv, blk256, 0, stream>>>(bufB, tc1_w, tc1_b, bufA);
    run_bn(bufA, tc1_g, tc1_be);
    conv1d_kernel<5><<<gConv, blk256, 0, stream>>>(bufA, tc2_w, tc2_b, bufB);
    run_bn(bufB, tc2_g, tc2_be);

    // --- LSTM x2 ---
    lstm_layer_kernel<<<BATCH, blk256, 0, stream>>>(bufB, wih0, whh0, bih0, bhh0, bufA);
    lstm_layer_kernel<<<BATCH, blk256, 0, stream>>>(bufA, wih1, whh1, bih1, bhh1, bufB);

    // --- Attention + head ---
    attn_logits_kernel<<<gRow, blk256, 0, stream>>>(bufB, attn_w, attn_b, logits);
    softmax_kernel<<<BATCH, blk256, 0, stream>>>(logits, attn_out);
    head_kernel<<<gRow, blk256, 0, stream>>>(bufB, attn_out,
                                             fc1_w, fc1_b, fc2_w, fc2_b, fc3_w, fc3_b,
                                             scores_out);
}
